// RNN_No_FFNN_88983132438935
// MI455X (gfx1250) — compile-verified, matrix operand order checked by reading, not run on hardware
//
#include <hip/hip_runtime.h>
#include <hip/hip_bf16.h>

// ---------------------------------------------------------------------------
// Problem constants (from reference)
// ---------------------------------------------------------------------------
constexpr int Bsz = 8192;   // batch
constexpr int Hd  = 2048;   // hidden
constexpr int Dd  = 104;    // line / output dim

// GEMM tiling
constexpr int BK   = 32;    // K-step = WMMA K for bf16
constexpr int LDSK = 40;    // padded LDS row stride in bf16 (80B, 16B-aligned)

typedef __attribute__((ext_vector_type(16))) __bf16 v16bf;
typedef __attribute__((ext_vector_type(8)))  float  v8f;

// ---------------------------------------------------------------------------
// Helpers
// ---------------------------------------------------------------------------
__device__ __forceinline__ v8f wmma_bf16(v16bf a, v16bf b, v8f c) {
    // D = A(16x32 bf16) * B(32x16 bf16) + C(16x16 f32)
    return __builtin_amdgcn_wmma_f32_16x16x32_bf16(
        /*neg_a=*/false, a, /*neg_b=*/false, b,
        /*c_mod=*/(short)0, c, /*reuse_a=*/false, /*reuse_b=*/false);
}

__device__ __forceinline__ void split_store(__bf16* hi, __bf16* lo, float x) {
    __bf16 h = (__bf16)x;               // top ~8 mantissa bits
    *hi = h;
    *lo = (__bf16)(x - (float)h);       // residual ~8 bits
}

// Register-staged tile prefetch: each of 256 threads owns 4 consecutive f32
// per 1024-element chunk of a ROWS x 32 tile.
template<int TOTAL>
struct Pref { float4 v[(TOTAL + 1023) / 1024]; };

template<int TOTAL>
__device__ __forceinline__ Pref<TOTAL> load_tile(const float* __restrict__ src, int ld,
                                                 int k0, int kmax, int rvalid, int tid) {
    Pref<TOTAL> t;
    constexpr int NV = (TOTAL + 1023) / 1024;
#pragma unroll
    for (int u = 0; u < NV; ++u) {
        int i = tid * 4 + u * 1024;
        float4 v = make_float4(0.f, 0.f, 0.f, 0.f);
        if (TOTAL % 1024 == 0 || i < TOTAL) {
            int r = i >> 5;          // /32
            int c = i & 31;
            if (r < rvalid) {
                const float* g = src + (size_t)r * ld + (k0 + c);
                if (k0 + c + 3 < kmax) {
                    v = *reinterpret_cast<const float4*>(g);   // 16B-aligned by construction
                } else {
                    if (k0 + c + 0 < kmax) v.x = g[0];
                    if (k0 + c + 1 < kmax) v.y = g[1];
                    if (k0 + c + 2 < kmax) v.z = g[2];
                    if (k0 + c + 3 < kmax) v.w = g[3];
                }
            }
        }
        t.v[u] = v;
    }
    return t;
}

template<int TOTAL>
__device__ __forceinline__ void store_tile(const Pref<TOTAL>& t,
                                           __bf16* hi, __bf16* lo, int tid) {
    constexpr int NV = (TOTAL + 1023) / 1024;
#pragma unroll
    for (int u = 0; u < NV; ++u) {
        int i = tid * 4 + u * 1024;
        if (TOTAL % 1024 == 0 || i < TOTAL) {
            int r = i >> 5;
            int c = i & 31;
            __bf16* ph = hi + r * LDSK + c;
            __bf16* pl = lo + r * LDSK + c;
            split_store(ph + 0, pl + 0, t.v[u].x);
            split_store(ph + 1, pl + 1, t.v[u].y);
            split_store(ph + 2, pl + 2, t.v[u].z);
            split_store(ph + 3, pl + 3, t.v[u].w);
        }
    }
}

// Build a 16x32 bf16 WMMA fragment from an LDS tile (row-major, stride LDSK).
// ISA 16-bit operand layout: lanes 0-15 hold row r, K {0..7} (V0-3) and
// {16..23} (V4-7); lanes 16-31 hold K {8..15} and {24..31}.
// => two aligned ds_load_b128 per lane.
__device__ __forceinline__ v16bf load_frag(const __bf16* base /* frag row 0 */, int lane) {
    int r    = lane & 15;
    int half = (lane >> 4) & 1;
    const char* p = reinterpret_cast<const char*>(base + r * LDSK);
    union { v16bf bf; uint4 u[2]; } f;
    f.u[0] = *reinterpret_cast<const uint4*>(p + half * 16);        // K 0..7 / 8..15
    f.u[1] = *reinterpret_cast<const uint4*>(p + 32 + half * 16);   // K 16..23 / 24..31
    return f.bf;
}

// ---------------------------------------------------------------------------
// Kernel A: h_t = sigmoid(hidden @ U_w^T + line @ W_w^T + U_b + W_b)
// Block tile 128(M) x 64(N); 8 waves, each 32x32 (2x2 WMMA subtiles).
// bf16x3: acc += Ahi*Bhi + Ahi*Blo + Alo*Bhi  (~16-bit mantissa accuracy)
// Double-buffered LDS; global loads for tile k+1 issued before the WMMAs of
// tile k and consumed after them (latency hidden behind matrix work).
// ---------------------------------------------------------------------------
constexpr int BM = 128;
constexpr int BN = 64;

__device__ __forceinline__ void compute_tile_A(const __bf16* AsH, const __bf16* AsL,
                                               const __bf16* BsH, const __bf16* BsL,
                                               int waveM, int waveN, int lane,
                                               v8f acc[2][2]) {
    v16bf ah[2], al[2], bh[2], bl[2];
#pragma unroll
    for (int s = 0; s < 2; ++s) {
        ah[s] = load_frag(AsH + (waveM * 32 + s * 16) * LDSK, lane);
        al[s] = load_frag(AsL + (waveM * 32 + s * 16) * LDSK, lane);
        bh[s] = load_frag(BsH + (waveN * 32 + s * 16) * LDSK, lane);
        bl[s] = load_frag(BsL + (waveN * 32 + s * 16) * LDSK, lane);
    }
#pragma unroll
    for (int mi = 0; mi < 2; ++mi)
#pragma unroll
        for (int ni = 0; ni < 2; ++ni) {
            acc[mi][ni] = wmma_bf16(ah[mi], bh[ni], acc[mi][ni]);
            acc[mi][ni] = wmma_bf16(ah[mi], bl[ni], acc[mi][ni]);
            acc[mi][ni] = wmma_bf16(al[mi], bh[ni], acc[mi][ni]);
        }
}

__global__ __launch_bounds__(256)
void rnn_gate_kernel(const float* __restrict__ hidden,
                     const float* __restrict__ line,
                     const float* __restrict__ U_w,
                     const float* __restrict__ W_w,
                     const float* __restrict__ U_b,
                     const float* __restrict__ W_b,
                     float* __restrict__ h_t) {
    __shared__ __bf16 AsH[2][BM * LDSK];
    __shared__ __bf16 AsL[2][BM * LDSK];
    __shared__ __bf16 BsH[2][BN * LDSK];
    __shared__ __bf16 BsL[2][BN * LDSK];

    const int tid   = threadIdx.x;
    const int lane  = tid & 31;
    const int wave  = tid >> 5;
    const int waveM = wave & 3;     // 0..3  -> M offset within block
    const int waveN = wave >> 2;    // 0..1  -> N offset within block
    const int m0 = blockIdx.y * BM;
    const int n0 = blockIdx.x * BN;

    v8f acc[2][2] = {};             // f32 accumulators, 16x16 each

    // ---- pass 1: hidden @ U_w^T  (K = Hd), software-pipelined ----
    const float* Abase = hidden + (size_t)m0 * Hd;
    const float* Bbase = U_w    + (size_t)n0 * Hd;   // C = A @ B^T (both K-contig)

    {
        Pref<BM * 32> pa = load_tile<BM * 32>(Abase, Hd, 0, Hd, BM, tid);
        Pref<BN * 32> pb = load_tile<BN * 32>(Bbase, Hd, 0, Hd, BN, tid);
        store_tile(pa, AsH[0], AsL[0], tid);
        store_tile(pb, BsH[0], BsL[0], tid);
    }
    __syncthreads();

    int cur = 0;
    for (int k0 = 0; k0 < Hd; k0 += BK) {
        const bool has_next = (k0 + BK) < Hd;
        Pref<BM * 32> na;
        Pref<BN * 32> nb;
        if (has_next) {   // issue global loads for tile k+1 (no wait yet)
            na = load_tile<BM * 32>(Abase, Hd, k0 + BK, Hd, BM, tid);
            nb = load_tile<BN * 32>(Bbase, Hd, k0 + BK, Hd, BN, tid);
        }
        compute_tile_A(AsH[cur], AsL[cur], BsH[cur], BsL[cur], waveM, waveN, lane, acc);
        if (has_next) {   // consume loads behind the WMMAs, fill other buffer
            store_tile(na, AsH[cur ^ 1], AsL[cur ^ 1], tid);
            store_tile(nb, BsH[cur ^ 1], BsL[cur ^ 1], tid);
        }
        __syncthreads();
        cur ^= 1;
    }

    // ---- pass 2: line @ W_w^T  (K = Dd = 104, zero-padded tail; ~5% of work) ----
    const float* A2 = line + (size_t)m0 * Dd;
    const float* B2 = W_w  + (size_t)n0 * Dd;
    for (int k0 = 0; k0 < Dd; k0 += BK) {
        __syncthreads();
        {
            Pref<BM * 32> qa = load_tile<BM * 32>(A2, Dd, k0, Dd, BM, tid);
            Pref<BN * 32> qb = load_tile<BN * 32>(B2, Dd, k0, Dd, BN, tid);
            store_tile(qa, AsH[0], AsL[0], tid);
            store_tile(qb, BsH[0], BsL[0], tid);
        }
        __syncthreads();
        compute_tile_A(AsH[0], AsL[0], BsH[0], BsL[0], waveM, waveN, lane, acc);
    }

    // ---- epilogue: +bias, sigmoid, store h_t ----
    // C/D layout: VGPR v, lanes 0-15 -> (M=v, N=lane); lanes 16-31 -> (M=v+8).
    const int cHalf = lane >> 4;
    const int cCol  = lane & 15;
#pragma unroll
    for (int mi = 0; mi < 2; ++mi)
#pragma unroll
        for (int ni = 0; ni < 2; ++ni) {
            int col  = n0 + waveN * 32 + ni * 16 + cCol;
            float bias = U_b[col] + W_b[col];
#pragma unroll
            for (int v = 0; v < 8; ++v) {
                int row = m0 + waveM * 32 + mi * 16 + v + 8 * cHalf;
                float x = acc[mi][ni][v] + bias;
                h_t[(size_t)row * Hd + col] = 1.0f / (1.0f + __expf(-x));
            }
        }
}

// ---------------------------------------------------------------------------
// Kernel B: pred = h_t @ V_w^T + V_b; segmented log-softmax over 104 cols.
// 128 rows / block; each wave owns 16 rows x 112 padded cols (7 subtiles).
// Same double-buffered pipeline; pred tile staged to LDS (overlapping the
// dead staging buffers), then one thread per row does the 10 segment
// log-softmaxes.
// ---------------------------------------------------------------------------
constexpr int BM2   = 128;
constexpr int NP    = 112;  // 104 padded to 7*16
constexpr int NSUB  = 7;
constexpr int PREDS = 113;  // padded pred row stride (odd dwords -> conflict-free)

constexpr int BUFB  = (BM2 * LDSK * 2 + NP * LDSK * 2) * (int)sizeof(__bf16); // 38400 B
constexpr int SMEMB = (2 * BUFB > BM2 * PREDS * (int)sizeof(float))
                        ? 2 * BUFB : BM2 * PREDS * (int)sizeof(float);        // 76800 B

__device__ __forceinline__ void compute_tile_B(const __bf16* Ah, const __bf16* Al,
                                               const __bf16* Bh, const __bf16* Bl,
                                               int wave, int lane, v8f acc[NSUB]) {
    v16bf ah = load_frag(Ah + wave * 16 * LDSK, lane);
    v16bf al = load_frag(Al + wave * 16 * LDSK, lane);
#pragma unroll
    for (int n = 0; n < NSUB; ++n) {
        v16bf bh = load_frag(Bh + n * 16 * LDSK, lane);
        v16bf bl = load_frag(Bl + n * 16 * LDSK, lane);
        acc[n] = wmma_bf16(ah, bh, acc[n]);
        acc[n] = wmma_bf16(ah, bl, acc[n]);
        acc[n] = wmma_bf16(al, bh, acc[n]);
    }
}

__global__ __launch_bounds__(256)
void head_logsoftmax_kernel(const float* __restrict__ h_t,
                            const float* __restrict__ V_w,
                            const float* __restrict__ V_b,
                            float* __restrict__ out) {
    __shared__ __align__(16) char smem[SMEMB];
    __bf16* Ah[2]; __bf16* Al[2]; __bf16* Bh[2]; __bf16* Bl[2];
#pragma unroll
    for (int b = 0; b < 2; ++b) {
        __bf16* base = reinterpret_cast<__bf16*>(smem + b * BUFB);
        Ah[b] = base;
        Al[b] = Ah[b] + BM2 * LDSK;
        Bh[b] = Al[b] + BM2 * LDSK;
        Bl[b] = Bh[b] + NP * LDSK;
    }
    float* pred = reinterpret_cast<float*>(smem);   // overlaps; used after k-loop

    const int tid  = threadIdx.x;
    const int lane = tid & 31;
    const int wave = tid >> 5;                      // wave w owns rows 16w..16w+15
    const int m0   = blockIdx.x * BM2;

    v8f acc[NSUB] = {};

    const float* Abase = h_t + (size_t)m0 * Hd;
    {
        Pref<BM2 * 32> pa = load_tile<BM2 * 32>(Abase, Hd, 0, Hd, BM2, tid);
        Pref<NP * 32>  pv = load_tile<NP * 32>(V_w,   Hd, 0, Hd, Dd,  tid); // rows>=104 zeroed
        store_tile(pa, Ah[0], Al[0], tid);
        store_tile(pv, Bh[0], Bl[0], tid);
    }
    __syncthreads();

    int cur = 0;
    for (int k0 = 0; k0 < Hd; k0 += BK) {
        const bool has_next = (k0 + BK) < Hd;
        Pref<BM2 * 32> na;
        Pref<NP * 32>  nv;
        if (has_next) {
            na = load_tile<BM2 * 32>(Abase, Hd, k0 + BK, Hd, BM2, tid);
            nv = load_tile<NP * 32>(V_w,    Hd, k0 + BK, Hd, Dd,  tid);
        }
        compute_tile_B(Ah[cur], Al[cur], Bh[cur], Bl[cur], wave, lane, acc);
        if (has_next) {
            store_tile(na, Ah[cur ^ 1], Al[cur ^ 1], tid);
            store_tile(nv, Bh[cur ^ 1], Bl[cur ^ 1], tid);
        }
        __syncthreads();
        cur ^= 1;
    }

    // write pred (+bias) tile into LDS (staging buffers are dead now)
    const int cHalf = lane >> 4;
    const int cCol  = lane & 15;
#pragma unroll
    for (int n = 0; n < NSUB; ++n) {
        int col = n * 16 + cCol;
        float bias = (col < Dd) ? V_b[col] : 0.f;
#pragma unroll
        for (int v = 0; v < 8; ++v) {
            int row = wave * 16 + v + 8 * cHalf;
            pred[row * PREDS + col] = acc[n][v] + bias;
        }
    }
    __syncthreads();

    // segmented log-softmax: one thread per row
    if (tid < BM2) {
        const int bounds[11] = {0, 13, 26, 39, 48, 52, 65, 78, 91, 100, 104};
        const float* pr = pred + tid * PREDS;
        float* orow = out + (size_t)(m0 + tid) * Dd;
        for (int s = 0; s < 10; ++s) {
            int a = bounds[s], b = bounds[s + 1];
            float mx = -3.0e38f;
            for (int j = a; j < b; ++j) mx = fmaxf(mx, pr[j]);
            float sum = 0.f;
            for (int j = a; j < b; ++j) sum += __expf(pr[j] - mx);
            float lse = mx + __logf(sum);
            for (int j = a; j < b; ++j) orow[j] = pr[j] - lse;
        }
    }
}

// ---------------------------------------------------------------------------
// Launch
// ---------------------------------------------------------------------------
extern "C" void kernel_launch(void* const* d_in, const int* in_sizes, int n_in,
                              void* d_out, int out_size, void* d_ws, size_t ws_size,
                              hipStream_t stream) {
    const float* line   = (const float*)d_in[0];   // [B, 104]
    const float* hidden = (const float*)d_in[1];   // [B, H]
    const float* U_w    = (const float*)d_in[2];   // [H, H]
    const float* U_b    = (const float*)d_in[3];   // [H]
    const float* W_w    = (const float*)d_in[4];   // [H, 104]
    const float* W_b    = (const float*)d_in[5];   // [H]
    const float* V_w    = (const float*)d_in[6];   // [104, H]
    const float* V_b    = (const float*)d_in[7];   // [104]

    float* out_ls = (float*)d_out;                 // [B, 104] log-softmax
    float* h_t    = out_ls + (size_t)Bsz * Dd;     // [B, H] gate output

    dim3 gridA(Hd / BN, Bsz / BM);    // (32, 64)
    rnn_gate_kernel<<<gridA, 256, 0, stream>>>(hidden, line, U_w, W_w, U_b, W_b, h_t);

    dim3 gridB(Bsz / BM2);            // 64
    head_logsoftmax_kernel<<<gridB, 256, 0, stream>>>(h_t, V_w, V_b, out_ls);
}
